// MambaLSTMProcessingBlock_8083128451707
// MI455X (gfx1250) — compile-verified, matrix operand order checked by reading, not run on hardware
//
#include <hip/hip_runtime.h>
#include <hip/hip_bf16.h>

// ---------------------------------------------------------------------------
// MambaLSTMProcessingBlock for MI455X (gfx1250, wave32, WMMA 16x16x32 bf16)
//
// Kernel 1: time branch (xLSTM)  - one block per (b,f) sequence, 256 blocks
// Kernel 2: freq branch (Mamba)  - one block per (b,t) sequence, 256 blocks
// All GEMMs via v_wmma_f32_16x16x32_bf16 (fp32 accumulate). Per-wave weight
// B-fragments preloaded into registers once. Sequential scans in registers,
// 16 steps per chunk. All fp32 divides -> v_rcp_f32; wave-uniform values
// forced to SGPRs via readfirstlane so gate selects are scalar branches.
// ---------------------------------------------------------------------------

typedef __attribute__((ext_vector_type(16))) __bf16 v16bf;
typedef __attribute__((ext_vector_type(8)))  float  v8f;

#define NT 128   // sequence length (T or F)
#define NC 64    // d_model
#define DIN 128  // d_inner
#define NSTATE 16
#define VZERO {0.f, 0.f, 0.f, 0.f, 0.f, 0.f, 0.f, 0.f}

__device__ __forceinline__ int kmap(int e, int half) {
  // 16-bit A/B fragment K mapping (CDNA5 ISA 7.12.2):
  // lanes 0-15 hold K = {0..7, 16..23}, lanes 16-31 hold K = {8..15, 24..31}
  return e + (e >= 8 ? 8 : 0) + (half ? 8 : 0);
}

// A fragment (16x32, bf16) from an f32 LDS tile, converting on the fly.
__device__ __forceinline__ v16bf load_a_f32(const float* src, int ld, int row0, int k0) {
  int lane = threadIdx.x & 31;
  int half = lane >> 4, m = lane & 15;
  const float* p = src + (row0 + m) * ld + k0;
  v16bf a;
#pragma unroll
  for (int e = 0; e < 16; ++e) a[e] = (__bf16)p[kmap(e, half)];
  return a;
}

// A fragment from a bf16 LDS tile.
__device__ __forceinline__ v16bf load_a_bf16(const __bf16* src, int ld, int row0, int k0) {
  int lane = threadIdx.x & 31;
  int half = lane >> 4, m = lane & 15;
  const __bf16* p = src + (row0 + m) * ld + k0;
  v16bf a;
#pragma unroll
  for (int e = 0; e < 16; ++e) a[e] = p[kmap(e, half)];
  return a;
}

// B fragment (32x16, bf16) for B = W^T, W row-major [N][K] fp32 in global.
__device__ __forceinline__ v16bf load_b_wt(const float* __restrict__ W, int K,
                                           int n0, int k0) {
  int lane = threadIdx.x & 31;
  int half = lane >> 4, n = lane & 15;
  const float* p = W + (n0 + n) * K + k0;
  v16bf b;
#pragma unroll
  for (int e = 0; e < 16; ++e) b[e] = (__bf16)p[kmap(e, half)];
  return b;
}

__device__ __forceinline__ v8f wmma_bf16(v16bf a, v16bf b, v8f c) {
  return __builtin_amdgcn_wmma_f32_16x16x32_bf16(false, a, false, b, (short)0, c,
                                                 false, false);
}

// Fast math: v_rcp_f32 instead of IEEE divide (adequate next to bf16 GEMMs).
__device__ __forceinline__ float rcp_(float x) { return __builtin_amdgcn_rcpf(x); }
__device__ __forceinline__ float sigmoidf_(float x) { return rcp_(1.f + __expf(-x)); }
__device__ __forceinline__ float siluf_(float x) { return x * sigmoidf_(x); }
// inf-safe fast tanh: t = exp(2|x|) -> 1 - 2/(t+1), restore sign
__device__ __forceinline__ float tanhf_fast(float x) {
  float t = __expf(2.f * fabsf(x));
  float r = 1.f - 2.f * rcp_(t + 1.f);
  return copysignf(r, x);
}
__device__ __forceinline__ float softplus_fast(float x) {
  return (x > 15.f) ? x : __logf(1.f + __expf(x));
}

// ---------------------------------------------------------------------------
// Kernel 1: time branch. grid = 256 (b*f), block = 256 threads (8 waves).
// ---------------------------------------------------------------------------
__global__ __launch_bounds__(256) void xlstm_time_kernel(
    const float* __restrict__ x, const float* __restrict__ ln_g,
    const float* __restrict__ ln_b, const float* __restrict__ w_o,
    const float* __restrict__ w_i, const float* __restrict__ w_f,
    const float* __restrict__ w_g, const float* __restrict__ w_out,
    float* __restrict__ x_time_out) {
  __shared__ float xn[NT][NC];        // layernormed input (residual + GEMM A)
  __shared__ float gates[4][16][NC];  // o,i,f,g post-activation, one chunk
  __shared__ float hbuf[16][NC];      // scan output chunk

  const int bf = blockIdx.x;
  const int b = bf >> 7, f = bf & 127;
  const int tid = threadIdx.x, lane = tid & 31;
  // wave id is wave-uniform: pin it to an SGPR so all derived selects/branches
  // are scalar (s_cbranch) instead of exec-mask manipulation.
  const int wave = __builtin_amdgcn_readfirstlane(tid >> 5);
  const int half = lane >> 4, col = lane & 15;

  // ---- per-wave weight B-fragments, loaded ONCE (chunk-invariant tiles)
  // tile = wave      -> gate gsel   (0: w_o, 1: w_i), n-tile = wave&3
  // tile = wave + 8  -> gate gsel+2 (2: w_f, 3: w_g), same n-tile
  const int gsel = wave >> 2, nt0 = wave & 3;
  v16bf bg0[2], bg1[2], bo[2];
  {
    const float* W0 = gsel ? w_i : w_o;
    const float* W1 = gsel ? w_g : w_f;
#pragma unroll
    for (int k2 = 0; k2 < 2; ++k2) {
      bg0[k2] = load_b_wt(W0, NC, nt0 * 16, k2 * 32);
      bg1[k2] = load_b_wt(W1, NC, nt0 * 16, k2 * 32);
    }
  }
  if (wave < 4) {
#pragma unroll
    for (int k2 = 0; k2 < 2; ++k2) bo[k2] = load_b_wt(w_out, NC, wave * 16, k2 * 32);
  }

  // ---- load + transpose x[b, c, t, f] -> xn[t][c]
  for (int idx = tid; idx < NT * NC; idx += 256) {
    int t = idx & 127, c = idx >> 7;
    xn[t][c] = x[(((b * 64 + c) * 128) + t) * 128 + f];
  }
  __syncthreads();

  // ---- LayerNorm over C per row
  if (tid < NT) {
    float mu = 0.f;
    for (int c = 0; c < NC; ++c) mu += xn[tid][c];
    mu *= (1.f / NC);
    float var = 0.f;
    for (int c = 0; c < NC; ++c) { float d = xn[tid][c] - mu; var += d * d; }
    float rstd = rsqrtf(var * (1.f / NC) + 1e-5f);
    for (int c = 0; c < NC; ++c)
      xn[tid][c] = (xn[tid][c] - mu) * rstd * ln_g[c] + ln_b[c];
  }
  __syncthreads();

  float cn = 0.f, cs = 0.f;  // xLSTM scan state (threads 0..63)

  for (int ch = 0; ch < 8; ++ch) {
    const int row0 = ch * 16;

    // --- gate GEMMs: one A fragment feeds both gate tiles of this wave
    v8f acc0 = VZERO, acc1 = VZERO;
#pragma unroll
    for (int k2 = 0; k2 < 2; ++k2) {
      v16bf a = load_a_f32(&xn[0][0], NC, row0, k2 * 32);
      acc0 = wmma_bf16(a, bg0[k2], acc0);
      acc1 = wmma_bf16(a, bg1[k2], acc1);
    }
    if (gsel == 0) {  // scalar branch: o-gate and f-gate
#pragma unroll
      for (int r = 0; r < 8; ++r) {
        gates[0][r + 8 * half][nt0 * 16 + col] = sigmoidf_(acc0[r]);
        gates[2][r + 8 * half][nt0 * 16 + col] = __expf(sigmoidf_(acc1[r]));
      }
    } else {          // i-gate and g-gate
#pragma unroll
      for (int r = 0; r < 8; ++r) {
        gates[1][r + 8 * half][nt0 * 16 + col] = __expf(acc0[r]);
        gates[3][r + 8 * half][nt0 * 16 + col] = tanhf_fast(acc1[r]);
      }
    }
    __syncthreads();

    // --- sequential scan, 16 steps, one thread per channel
    if (tid < NC) {
      int c = tid;
      for (int s = 0; s < 16; ++s) {
        float fe = gates[2][s][c], ii = gates[1][s][c];
        cn = fe * cn + ii * gates[3][s][c];
        cs = fe * cs + ii;
        hbuf[s][c] = gates[0][s][c] * cn * rcp_(cs);
      }
    }
    __syncthreads();

    // --- out GEMM: h_chunk @ w_out^T (waves 0..3), + residual
    if (wave < 4) {
      v8f acc = VZERO;
#pragma unroll
      for (int k2 = 0; k2 < 2; ++k2)
        acc = wmma_bf16(load_a_f32(&hbuf[0][0], NC, 0, k2 * 32), bo[k2], acc);
#pragma unroll
      for (int r = 0; r < 8; ++r) {
        int t = row0 + r + 8 * half;
        int c2 = wave * 16 + col;
        x_time_out[(bf * 128 + t) * 64 + c2] = acc[r] + xn[t][c2];
      }
    }
    __syncthreads();
  }
}

// ---------------------------------------------------------------------------
// Kernel 2: freq branch (Mamba). grid = 256 (b*t), block = 256 threads.
// ---------------------------------------------------------------------------
__global__ __launch_bounds__(256) void mamba_freq_kernel(
    const float* __restrict__ xin, const float* __restrict__ x_orig,
    const float* __restrict__ ln_g, const float* __restrict__ ln_b,
    const float* __restrict__ m_in,   // [256][64]
    const float* __restrict__ convw,  // [128][4]
    const float* __restrict__ convb,  // [128]
    const float* __restrict__ m_b,    // [16][128]
    const float* __restrict__ m_c,    // [16][128]
    const float* __restrict__ m_dtw,  // [128][128]
    const float* __restrict__ m_dtb,  // [128]
    const float* __restrict__ A_log,  // [128][16]
    const float* __restrict__ Dp,     // [128]
    const float* __restrict__ m_out,  // [64][128]
    float* __restrict__ out) {
  __shared__ float  xn[NT][NC];        // layernormed seq (residual + GEMM A)
  __shared__ float  xs_buf[19][DIN];   // conv input: 3-row history + 16 rows
  __shared__ __bf16 res_b[16][DIN];    // gate branch of in-proj
  __shared__ float  u_c[16][DIN];      // conv+silu output; reused as staging
  __shared__ __bf16 dtp_b[16][DIN];    // delta pre-activation
  __shared__ float  bmat[16][NSTATE];
  __shared__ float  cmat[16][NSTATE];
  __shared__ __bf16 zb[16][DIN];       // (y + u*D) * silu(res)

  const int bt = blockIdx.x;
  const int b = bt >> 7, t = bt & 127;
  const int tid = threadIdx.x, lane = tid & 31;
  const int wave = __builtin_amdgcn_readfirstlane(tid >> 5);
  const int half = lane >> 4, col = lane & 15;

  // ---- per-wave weight B-fragments, loaded ONCE
  v16bf bi0[2], bi1[2];  // m_in tiles wave (-> xs) and wave+8 (-> res)
#pragma unroll
  for (int k2 = 0; k2 < 2; ++k2) {
    bi0[k2] = load_b_wt(m_in, NC, wave * 16, k2 * 32);
    bi1[k2] = load_b_wt(m_in, NC, (wave + 8) * 16, k2 * 32);
  }
  v16bf bdt[4];          // m_dtw tile wave
#pragma unroll
  for (int k4 = 0; k4 < 4; ++k4) bdt[k4] = load_b_wt(m_dtw, DIN, wave * 16, k4 * 32);
  v16bf bbc[4];          // wave 6 -> m_b, wave 7 -> m_c
  if (wave >= 6) {
    const float* W = (wave == 6) ? m_b : m_c;
#pragma unroll
    for (int k4 = 0; k4 < 4; ++k4) bbc[k4] = load_b_wt(W, DIN, 0, k4 * 32);
  }
  v16bf bout[4];         // m_out tile wave (waves 0..3)
  if (wave < 4) {
#pragma unroll
    for (int k4 = 0; k4 < 4; ++k4) bout[k4] = load_b_wt(m_out, DIN, wave * 16, k4 * 32);
  }

  // ---- load x_freq[p][c] = ws[(b*128 + p)][t][c] (coalesced along c)
  for (int idx = tid; idx < NT * NC; idx += 256) {
    int p = idx >> 6, c = idx & 63;
    xn[p][c] = xin[((b * 128 + p) * 128 + t) * 64 + c];
  }
  __syncthreads();

  if (tid < NT) {
    float mu = 0.f;
    for (int c = 0; c < NC; ++c) mu += xn[tid][c];
    mu *= (1.f / NC);
    float var = 0.f;
    for (int c = 0; c < NC; ++c) { float d = xn[tid][c] - mu; var += d * d; }
    float rstd = rsqrtf(var * (1.f / NC) + 1e-5f);
    for (int c = 0; c < NC; ++c)
      xn[tid][c] = (xn[tid][c] - mu) * rstd * ln_g[c] + ln_b[c];
  }
  for (int idx = tid; idx < 3 * DIN; idx += 256) xs_buf[idx >> 7][idx & 127] = 0.f;
  __syncthreads();

  // ---- per-thread scan state: pair (d = tid>>1) splits the 16 states
  const int d = tid >> 1, halfn = tid & 1;
  float Areg[8], hstate[8];
#pragma unroll
  for (int n = 0; n < 8; ++n) {
    Areg[n] = -__expf(A_log[d * NSTATE + halfn * 8 + n]);
    hstate[n] = 0.f;
  }
  const float Dreg = Dp[d];
  const float dtb = m_dtb[d];
  // conv taps: each thread always handles channel dc = tid & 127
  const int dc = tid & 127;
  float cw0 = convw[dc * 4 + 0], cw1 = convw[dc * 4 + 1];
  float cw2 = convw[dc * 4 + 2], cw3 = convw[dc * 4 + 3];
  float cb = convb[dc];

  for (int ch = 0; ch < 8; ++ch) {
    const int row0 = ch * 16;

    // --- in-projection: shared A fragment, two N-tiles per wave
    {
      v8f a0 = VZERO, a1 = VZERO;
#pragma unroll
      for (int k2 = 0; k2 < 2; ++k2) {
        v16bf a = load_a_f32(&xn[0][0], NC, row0, k2 * 32);
        a0 = wmma_bf16(a, bi0[k2], a0);
        a1 = wmma_bf16(a, bi1[k2], a1);
      }
#pragma unroll
      for (int r = 0; r < 8; ++r) {
        int rr = r + 8 * half;
        xs_buf[3 + rr][wave * 16 + col] = a0[r];
        res_b[rr][wave * 16 + col] = (__bf16)a1[r];
      }
    }
    __syncthreads();

    // --- causal depthwise conv (k=4) + SiLU (registers hold taps)
    for (int idx = tid; idx < 16 * DIN; idx += 256) {
      int p = idx >> 7;  // channel = dc (constant per thread)
      float s = cb + cw0 * xs_buf[p + 0][dc] + cw1 * xs_buf[p + 1][dc] +
                cw2 * xs_buf[p + 2][dc] + cw3 * xs_buf[p + 3][dc];
      u_c[p][dc] = siluf_(s);
    }
    __syncthreads();

    // slide conv history (rows 16..18 -> 0..2) for next chunk
    for (int idx = tid; idx < 3 * DIN; idx += 256)
      xs_buf[idx >> 7][idx & 127] = xs_buf[16 + (idx >> 7)][idx & 127];

    // --- delta GEMM (all waves) + B/C GEMMs (waves 6,7)
    {
      v8f acc = VZERO;
#pragma unroll
      for (int k4 = 0; k4 < 4; ++k4)
        acc = wmma_bf16(load_a_f32(&u_c[0][0], DIN, 0, k4 * 32), bdt[k4], acc);
#pragma unroll
      for (int r = 0; r < 8; ++r)
        dtp_b[r + 8 * half][wave * 16 + col] = (__bf16)acc[r];
    }
    if (wave >= 6) {
      v8f acc = VZERO;
#pragma unroll
      for (int k4 = 0; k4 < 4; ++k4)
        acc = wmma_bf16(load_a_f32(&u_c[0][0], DIN, 0, k4 * 32), bbc[k4], acc);
#pragma unroll
      for (int r = 0; r < 8; ++r) {
        if (wave == 6) bmat[r + 8 * half][col] = acc[r];
        else           cmat[r + 8 * half][col] = acc[r];
      }
    }
    __syncthreads();

    // --- selective scan, 16 steps; lane pairs combine their 8-state halves
    for (int s = 0; s < 16; ++s) {
      float delta = softplus_fast((float)dtp_b[s][d] + dtb);
      float uv = u_c[s][d];
      float y = 0.f;
#pragma unroll
      for (int n = 0; n < 8; ++n) {
        float dA = __expf(delta * Areg[n]);
        hstate[n] = dA * hstate[n] + delta * bmat[s][halfn * 8 + n] * uv;
        y += hstate[n] * cmat[s][halfn * 8 + n];
      }
      y += __shfl_xor(y, 1, 32);  // combine the two halves (wave32 pair)
      if (halfn == 0) {
        float yy = y + uv * Dreg;
        zb[s][d] = (__bf16)(yy * siluf_((float)res_b[s][d]));
      }
    }
    __syncthreads();

    // --- out GEMM: z @ m_out^T -> stage to LDS (u_c is dead now)
    float* stg = &u_c[0][0];  // [16][64] staging, row-major
    if (wave < 4) {
      v8f acc = VZERO;
#pragma unroll
      for (int k4 = 0; k4 < 4; ++k4)
        acc = wmma_bf16(load_a_bf16(&zb[0][0], DIN, 0, k4 * 32), bout[k4], acc);
#pragma unroll
      for (int r = 0; r < 8; ++r) {
        int pl = r + 8 * half;        // local frequency position
        int c2 = wave * 16 + col;     // channel
        stg[pl * 64 + c2] = acc[r] + xn[row0 + pl][c2];
      }
    }
    __syncthreads();

    // --- coalesced final store: lanes run along p (contiguous in memory)
    for (int idx = tid; idx < 16 * 64; idx += 256) {
      int c2 = idx >> 4, pl = idx & 15;
      int oi = ((b * 64 + c2) * 128 + t) * 128 + (row0 + pl);
      out[oi] = stg[pl * 64 + c2] + x_orig[oi];
    }
    __syncthreads();
  }
}

// ---------------------------------------------------------------------------
extern "C" void kernel_launch(void* const* d_in, const int* in_sizes, int n_in,
                              void* d_out, int out_size, void* d_ws, size_t ws_size,
                              hipStream_t stream) {
  const float* x      = (const float*)d_in[0];
  const float* ln1_g  = (const float*)d_in[1];
  const float* ln1_b  = (const float*)d_in[2];
  const float* ln2_g  = (const float*)d_in[3];
  const float* ln2_b  = (const float*)d_in[4];
  const float* xw_o   = (const float*)d_in[5];
  const float* xw_i   = (const float*)d_in[6];
  const float* xw_f   = (const float*)d_in[7];
  const float* xw_g   = (const float*)d_in[8];
  const float* xw_out = (const float*)d_in[9];
  const float* m_in   = (const float*)d_in[10];
  const float* m_cw   = (const float*)d_in[11];
  const float* m_cb   = (const float*)d_in[12];
  const float* m_b    = (const float*)d_in[13];
  const float* m_c    = (const float*)d_in[14];
  const float* m_dtw  = (const float*)d_in[15];
  const float* m_dtb  = (const float*)d_in[16];
  const float* m_Alog = (const float*)d_in[17];
  const float* m_D    = (const float*)d_in[18];
  const float* m_out  = (const float*)d_in[19];

  float* ws = (float*)d_ws;  // time-branch output: 256*128*64 fp32 = 8 MB

  xlstm_time_kernel<<<256, 256, 0, stream>>>(x, ln1_g, ln1_b, xw_o, xw_i, xw_f,
                                             xw_g, xw_out, ws);
  mamba_freq_kernel<<<256, 256, 0, stream>>>(ws, x, ln2_g, ln2_b, m_in, m_cw,
                                             m_cb, m_b, m_c, m_dtw, m_dtb,
                                             m_Alog, m_D, m_out, (float*)d_out);
}
